// MultiHeadFinancialAttention_30709016166628
// MI455X (gfx1250) — compile-verified, matrix operand order checked by reading, not run on hardware
//
#include <hip/hip_runtime.h>
#include <hip/hip_bf16.h>
#include <cmath>

#define D_MODEL 1024
#define N_HEADS 16
#define D_KH    64
#define SEQ     2048
#define BATCH   2
#define MROWS   (BATCH * SEQ)     // 4096

typedef __attribute__((ext_vector_type(16))) _Float16     v16h;
typedef __attribute__((ext_vector_type(8)))  float        v8f;
typedef __attribute__((ext_vector_type(4)))  unsigned int v4u;
typedef __attribute__((ext_vector_type(8)))  int          v8i;
typedef __attribute__((ext_vector_type(4)))  int          v4i;

__device__ __forceinline__ v8f wmma_f16(v16h a, v16h b, v8f c) {
    return __builtin_amdgcn_wmma_f32_16x16x32_f16(
        false, a, false, b, (short)0, c, false, false);
}

// Low 32 bits of a generic pointer to LDS == LDS byte offset.
__device__ __forceinline__ unsigned lds_off(const void* p) {
    return (unsigned)(size_t)p;
}

// ---------------------------------------------------------------------------
// TDM 2D tile load: tile_x elems (contiguous) x tile_y rows, row stride
// `stride` elems, element size 2^dsize bytes, into LDS with a 1-DWORD pad
// every 32 DWORDs (so tile rows must be 128B wide for the pad to land at
// row boundaries).  6-arg clang-23 builtin form.
// ---------------------------------------------------------------------------
__device__ __forceinline__ void tdm_load_2d(unsigned lds, const void* gptr,
                                            unsigned dsize, unsigned tile_x,
                                            unsigned tile_y, unsigned stride,
                                            unsigned dim0, unsigned dim1) {
    const unsigned long long ga = (unsigned long long)(size_t)gptr;
    v4u g0;
    g0[0] = 1u;                                           // count=1 (valid)
    g0[1] = lds;                                          // lds_addr
    g0[2] = (unsigned)(ga & 0xffffffffu);                 // global_addr[95:64]
    g0[3] = (unsigned)((ga >> 32) & 0x01ffffffu)          // global_addr[120:96]
          | (2u << 30);                                   // type=2 (image)
    v8i g1;
    g1[0] = (int)((dsize << 16)   // data_size code (1=2B, 2=4B)
          |       (1u << 20)      // pad_enable
          |       (4u << 22));    // pad_interval = 32 DWORDs (amount = 1 DWORD)
    g1[1] = (int)((dim0 & 0xffffu) << 16);                // tensor_dim0 lo
    g1[2] = (int)((dim0 >> 16) | ((dim1 & 0xffffu) << 16));
    g1[3] = (int)((dim1 >> 16) | (tile_x << 16));         // tile_dim0
    g1[4] = (int)(tile_y & 0xffffu);                      // tile_dim1 (dim2=0)
    g1[5] = (int)stride;                                  // tensor_dim0_stride
    g1[6] = 0;
    g1[7] = 0;
    const v4i z4 = {0, 0, 0, 0};
    const v8i z8 = {0, 0, 0, 0, 0, 0, 0, 0};
    __builtin_amdgcn_tensor_load_to_lds(g0, g1, z4, z4, z8, 0);
}

// ---------------------------------------------------------------------------
// Projection GEMM: Y[4096,1024] = X[4096,1024] @ W[1024,1024]^T + bias.
// 64x64 block tile, k-step 32, TDM double-buffered f32 staging in LDS,
// f32->f16 conversion at fragment read, 2 WMMAs per wave per k-step.
// TOUT = _Float16 for the Q/K/V projections, float for the output proj.
// ---------------------------------------------------------------------------
template <typename TOUT>
__global__ __launch_bounds__(256)
void proj_gemm_kernel(const float* __restrict__ X,
                      const float* __restrict__ W,
                      const float* __restrict__ bias,
                      TOUT* __restrict__ Y)
{
    __shared__ float As[2][64][33];   // 33-dw pitch from TDM pad
    __shared__ float Bs[2][64][33];

    const int tid  = threadIdx.x;
    const int lane = tid & 31;
    const int wv   = tid >> 5;
    const int half = lane >> 4;
    const int ln   = lane & 15;

    const int m0  = blockIdx.y * 64;
    const int n0  = blockIdx.x * 64;
    const int mt  = wv & 3;
    const int nt0 = (wv >> 2) * 2;

    v8f acc0 = {}; v8f acc1 = {};

    if (tid < 32) {   // prologue: TDM-load k0=0 tiles into buffer 0
        tdm_load_2d(lds_off(&As[0][0][0]), X + (size_t)m0 * D_MODEL,
                    2, 32, 64, D_MODEL, D_MODEL, MROWS);
        tdm_load_2d(lds_off(&Bs[0][0][0]), W + (size_t)n0 * D_MODEL,
                    2, 32, 64, D_MODEL, D_MODEL, D_MODEL);
    }

    int p = 0;
    for (int k0 = 0; k0 < D_MODEL; k0 += 32) {
        if (tid < 32) {
            if (k0 + 32 < D_MODEL) {   // issue next tile into other buffer
                tdm_load_2d(lds_off(&As[1 - p][0][0]),
                            X + (size_t)m0 * D_MODEL + k0 + 32,
                            2, 32, 64, D_MODEL, D_MODEL, MROWS);
                tdm_load_2d(lds_off(&Bs[1 - p][0][0]),
                            W + (size_t)n0 * D_MODEL + k0 + 32,
                            2, 32, 64, D_MODEL, D_MODEL, D_MODEL);
                __builtin_amdgcn_s_wait_tensorcnt(2);  // current buffer done
            } else {
                __builtin_amdgcn_s_wait_tensorcnt(0);
            }
        }
        __syncthreads();

        v16h a, b0, b1;
        {   // A fragment: row = mt*16+ln, K(e) = half*8+e | 16+half*8+e
            const float* ar = &As[p][mt * 16 + ln][half * 8];
            #pragma unroll
            for (int e = 0; e < 8; ++e) a[e]     = (_Float16)ar[e];
            #pragma unroll
            for (int e = 0; e < 8; ++e) a[8 + e] = (_Float16)ar[16 + e];
        }
        {
            const float* br = &Bs[p][nt0 * 16 + ln][half * 8];
            #pragma unroll
            for (int e = 0; e < 8; ++e) b0[e]     = (_Float16)br[e];
            #pragma unroll
            for (int e = 0; e < 8; ++e) b0[8 + e] = (_Float16)br[16 + e];
            br = &Bs[p][(nt0 + 1) * 16 + ln][half * 8];
            #pragma unroll
            for (int e = 0; e < 8; ++e) b1[e]     = (_Float16)br[e];
            #pragma unroll
            for (int e = 0; e < 8; ++e) b1[8 + e] = (_Float16)br[16 + e];
        }
        acc0 = wmma_f16(a, b0, acc0);
        acc1 = wmma_f16(a, b1, acc1);
        __syncthreads();
        p ^= 1;
    }

    // Epilogue: bias + store.  D layout: row = r + 8*half, col = ln.
    const int na = n0 + nt0 * 16 + ln;
    const float ba = bias[na];
    const float bb = bias[na + 16];
    TOUT* ya = Y + (size_t)(m0 + mt * 16 + 8 * half) * D_MODEL + na;
    #pragma unroll
    for (int r = 0; r < 8; ++r) {
        ya[(size_t)r * D_MODEL]      = (TOUT)(acc0[r] + ba);
        ya[(size_t)r * D_MODEL + 16] = (TOUT)(acc1[r] + bb);
    }
}

// ---------------------------------------------------------------------------
// Flash attention with additive time-decay bias.  Q/K/V are f16 (written by
// the projection GEMMs), ctx is f32.
// Block = 8 waves, all on the same (b,h), covering 128 consecutive q rows.
// Per 32-key step: TDM loads the K tile (f16, async, overlapping the manual
// transposed V staging); each wave: 4 score WMMAs, online softmax, P restage
// through per-wave LDS, 4 ctx WMMAs.
// ---------------------------------------------------------------------------
__global__ __launch_bounds__(256)
void attn_flash_kernel(const _Float16* __restrict__ Qb,
                       const _Float16* __restrict__ Kb,
                       const _Float16* __restrict__ Vb,
                       const float* __restrict__ td_ptr,
                       float* __restrict__ Cb)
{
    __shared__ _Float16 Kt[32][66];     // [key][d], 66-elem pitch from TDM pad
    __shared__ _Float16 Vt[64][34];     // [d][key] (+2 pad)
    __shared__ _Float16 P[8][16][32];   // per-wave probability tile

    const int tid  = threadIdx.x;
    const int lane = tid & 31;
    const int wv   = tid >> 5;
    const int half = lane >> 4;
    const int ln   = lane & 15;

    const int bh   = blockIdx.x >> 4;          // 0..31
    const int qblk = blockIdx.x & 15;          // 0..15
    const int b    = bh >> 4;
    const int h    = bh & 15;
    const int q0   = (qblk * 8 + wv) * 16;

    const float td    = td_ptr[0];
    const float scale = 0.125f;                // 1/sqrt(64)

    const _Float16* Qh = Qb + (size_t)b * SEQ * D_MODEL + h * D_KH;
    const _Float16* Kh = Kb + (size_t)b * SEQ * D_MODEL + h * D_KH;
    const _Float16* Vh = Vb + (size_t)b * SEQ * D_MODEL + h * D_KH;
    float*          Ch = Cb + (size_t)b * SEQ * D_MODEL + h * D_KH;

    // Q A-fragments for d = 0..31 / 32..63 (row = ln); contiguous f16 loads.
    v16h aq0, aq1;
    {
        const _Float16* qr = Qh + (size_t)(q0 + ln) * D_MODEL;
        #pragma unroll
        for (int e = 0; e < 8; ++e) aq0[e]     = qr[half * 8 + e];
        #pragma unroll
        for (int e = 0; e < 8; ++e) aq0[8 + e] = qr[16 + half * 8 + e];
        #pragma unroll
        for (int e = 0; e < 8; ++e) aq1[e]     = qr[32 + half * 8 + e];
        #pragma unroll
        for (int e = 0; e < 8; ++e) aq1[8 + e] = qr[48 + half * 8 + e];
    }

    v8f c0 = {}, c1 = {}, c2 = {}, c3 = {};
    float mrow[8], lrow[8];
    #pragma unroll
    for (int r = 0; r < 8; ++r) { mrow[r] = -3.0e38f; lrow[r] = 0.0f; }

    const int srow = tid >> 3;          // staging: key row 0..31
    const int scol = (tid & 7) * 8;     // staging: d col 0,8,..,56

    for (int key0 = 0; key0 < SEQ; key0 += 32) {
        // ---- async TDM load of the K tile (f16, 32 keys x 64 d) -----------
        if (tid < 32) {
            tdm_load_2d(lds_off(&Kt[0][0]), Kh + (size_t)key0 * D_MODEL,
                        1, 64, 32, D_MODEL, D_MODEL, SEQ);
        }
        // ---- manual transposed staging of the V tile (overlaps the TDM) ---
        {
            const _Float16* vs = Vh + (size_t)(key0 + srow) * D_MODEL + scol;
            if (key0 + 32 < SEQ)
                __builtin_prefetch(Vh + (size_t)(key0 + 32 + srow) * D_MODEL + scol, 0, 0);
            #pragma unroll
            for (int j = 0; j < 8; ++j) Vt[scol + j][srow] = vs[j];
        }
        if (tid < 32) __builtin_amdgcn_s_wait_tensorcnt(0);
        __syncthreads();

        // ---- scores: two 16x16 tiles from LDS K ---------------------------
        v8f s0 = {}, s1 = {};
        {
            v16h bk;
            const _Float16* kr = &Kt[ln][half * 8];
            #pragma unroll
            for (int e = 0; e < 8; ++e) bk[e]     = kr[e];
            #pragma unroll
            for (int e = 0; e < 8; ++e) bk[8 + e] = kr[16 + e];
            s0 = wmma_f16(aq0, bk, s0);
            #pragma unroll
            for (int e = 0; e < 8; ++e) bk[e]     = kr[32 + e];
            #pragma unroll
            for (int e = 0; e < 8; ++e) bk[8 + e] = kr[48 + e];
            s0 = wmma_f16(aq1, bk, s0);

            kr = &Kt[16 + ln][half * 8];
            #pragma unroll
            for (int e = 0; e < 8; ++e) bk[e]     = kr[e];
            #pragma unroll
            for (int e = 0; e < 8; ++e) bk[8 + e] = kr[16 + e];
            s1 = wmma_f16(aq0, bk, s1);
            #pragma unroll
            for (int e = 0; e < 8; ++e) bk[e]     = kr[32 + e];
            #pragma unroll
            for (int e = 0; e < 8; ++e) bk[8 + e] = kr[48 + e];
            s1 = wmma_f16(aq1, bk, s1);
        }

        // ---- scale + time-decay bias + online softmax ---------------------
        float alpha[8];
        #pragma unroll
        for (int r = 0; r < 8; ++r) {
            const float qpos = (float)(q0 + r + 8 * half);
            const float v0 = s0[r] * scale - td * fabsf(qpos - (float)(key0 + ln));
            const float v1 = s1[r] * scale - td * fabsf(qpos - (float)(key0 + 16 + ln));
            float t = fmaxf(v0, v1);
            t = fmaxf(t, __shfl_xor(t, 1, 32));
            t = fmaxf(t, __shfl_xor(t, 2, 32));
            t = fmaxf(t, __shfl_xor(t, 4, 32));
            t = fmaxf(t, __shfl_xor(t, 8, 32));
            const float mnew = fmaxf(mrow[r], t);
            const float a    = __expf(mrow[r] - mnew);
            const float p0   = __expf(v0 - mnew);
            const float p1   = __expf(v1 - mnew);
            float rs = p0 + p1;
            rs += __shfl_xor(rs, 1, 32);
            rs += __shfl_xor(rs, 2, 32);
            rs += __shfl_xor(rs, 4, 32);
            rs += __shfl_xor(rs, 8, 32);
            lrow[r]  = lrow[r] * a + rs;
            mrow[r]  = mnew;
            alpha[r] = a;
            P[wv][r + 8 * half][ln]      = (_Float16)p0;
            P[wv][r + 8 * half][16 + ln] = (_Float16)p1;
        }
        asm volatile("s_wait_dscnt 0x0" ::: "memory");   // LDS RAW fence (P)

        v16h ap;   // P as A fragment
        {
            const _Float16* pr = &P[wv][ln][half * 8];
            #pragma unroll
            for (int e = 0; e < 8; ++e) ap[e]     = pr[e];
            #pragma unroll
            for (int e = 0; e < 8; ++e) ap[8 + e] = pr[16 + e];
        }

        #pragma unroll
        for (int r = 0; r < 8; ++r) {
            c0[r] *= alpha[r]; c1[r] *= alpha[r];
            c2[r] *= alpha[r]; c3[r] *= alpha[r];
        }

        // ---- ctx += P(16x32) @ V(32x64) from transposed LDS V -------------
        {
            v16h bv;
            #pragma unroll
            for (int t = 0; t < 4; ++t) {
                const _Float16* vrp = &Vt[t * 16 + ln][half * 8];
                #pragma unroll
                for (int e = 0; e < 8; ++e) bv[e]     = vrp[e];
                #pragma unroll
                for (int e = 0; e < 8; ++e) bv[8 + e] = vrp[16 + e];
                if      (t == 0) c0 = wmma_f16(ap, bv, c0);
                else if (t == 1) c1 = wmma_f16(ap, bv, c1);
                else if (t == 2) c2 = wmma_f16(ap, bv, c2);
                else             c3 = wmma_f16(ap, bv, c3);
            }
        }
        __syncthreads();
    }

    // ---- epilogue: normalize and store ctx (f32) --------------------------
    #pragma unroll
    for (int r = 0; r < 8; ++r) {
        const float inv = 1.0f / lrow[r];
        float* cr = Ch + (size_t)(q0 + r + 8 * half) * D_MODEL + ln;
        cr[0]  = c0[r] * inv;
        cr[16] = c1[r] * inv;
        cr[32] = c2[r] * inv;
        cr[48] = c3[r] * inv;
    }
}

// ---------------------------------------------------------------------------
extern "C" void kernel_launch(void* const* d_in, const int* in_sizes, int n_in,
                              void* d_out, int out_size, void* d_ws, size_t ws_size,
                              hipStream_t stream) {
    const float* query = (const float*)d_in[0];
    const float* key   = (const float*)d_in[1];
    const float* value = (const float*)d_in[2];
    const float* wq = (const float*)d_in[3];
    const float* bq = (const float*)d_in[4];
    const float* wk = (const float*)d_in[5];
    const float* bk = (const float*)d_in[6];
    const float* wv = (const float*)d_in[7];
    const float* bv = (const float*)d_in[8];
    const float* wo = (const float*)d_in[9];
    const float* bo = (const float*)d_in[10];
    const float* td = (const float*)d_in[11];
    float* out = (float*)d_out;

    const size_t elems = (size_t)MROWS * D_MODEL;
    _Float16* Qbuf = (_Float16*)d_ws;            // 8 MB
    _Float16* Kbuf = Qbuf + elems;               // 8 MB
    _Float16* Vbuf = Kbuf + elems;               // 8 MB
    float*    Cbuf = (float*)(Vbuf + elems);     // 16 MB

    const dim3 gblk(256);
    const dim3 ggrid(D_MODEL / 64, MROWS / 64);  // 16 x 64
    proj_gemm_kernel<_Float16><<<ggrid, gblk, 0, stream>>>(query, wq, bq, Qbuf);
    proj_gemm_kernel<_Float16><<<ggrid, gblk, 0, stream>>>(key,   wk, bk, Kbuf);
    proj_gemm_kernel<_Float16><<<ggrid, gblk, 0, stream>>>(value, wv, bv, Vbuf);

    attn_flash_kernel<<<dim3(BATCH * N_HEADS * (SEQ / 128)), dim3(256), 0, stream>>>(
        Qbuf, Kbuf, Vbuf, td, Cbuf);

    proj_gemm_kernel<float><<<ggrid, gblk, 0, stream>>>(Cbuf, wo, bo, out);
}
